// DeformableTransformerDecoderLayer_77902116815218
// MI455X (gfx1250) — matrix operand order repaired and checked by reading, not compiled, not run on hardware
//
#include <hip/hip_runtime.h>
#include <math.h>

// ---------------- problem constants ----------------
#define Dd   256
#define Hh   8
#define Ll   4
#define Pp   4
#define NQn  900
#define Bb   4
#define DFFn 1024
#define Ss   21760   // 128*128 + 64*64 + 32*32 + 16*16
#define HDn  32

typedef __attribute__((ext_vector_type(16))) __bf16 bf16x16;
typedef __attribute__((ext_vector_type(8)))  __bf16 bf16x8;
typedef __attribute__((ext_vector_type(8)))  float  floatx8;

__device__ __forceinline__ bf16x8 pack_bf16x8(float4 a, float4 b)
{
    bf16x8 r;
    r[0] = (__bf16)a.x; r[1] = (__bf16)a.y; r[2] = (__bf16)a.z; r[3] = (__bf16)a.w;
    r[4] = (__bf16)b.x; r[5] = (__bf16)b.y; r[6] = (__bf16)b.z; r[7] = (__bf16)b.w;
    return r;
}

__device__ __forceinline__ bf16x16 cat16(bf16x8 lo, bf16x8 hi)
{
    return __builtin_shufflevector(lo, hi, 0, 1, 2, 3, 4, 5, 6, 7, 8, 9, 10, 11, 12, 13, 14, 15);
}

// =====================================================================
// Generic GEMM:  C[M,N] = act(A[M,K] * W[K,N] + bias)   (fp32 in/out,
// bf16 WMMA compute, fp32 accumulate).
// Block tile 128x64x32, 256 threads (8 waves), each wave computes a
// 32x32 sub-tile = 4 C-fragments = 4 WMMAs per K-step.
// LDS double buffer + register staging: one barrier per K-step;
// global loads of tile i+1 issue before the WMMAs of tile i.
// Running pointers: one 64-bit add per pointer per K-step.
// A in LDS row-major [BM][40]; B in LDS TRANSPOSED [BN][40] so every
// WMMA fragment is 2 aligned 16-byte ds_load_b128.
// =====================================================================
#define BM 128
#define BN 64
#define BK 32
#define LDSD 40   // padded row stride (elements); 80B: 16B-aligned, bank-friendly

__global__ __launch_bounds__(256) void gemm_bf16_wmma(
    const float* __restrict__ A, int M, int K, int lda,
    const float* __restrict__ W, int ldw,
    const float* __restrict__ bias,
    float* __restrict__ C, int N, int ldc, int relu)
{
    __shared__ __attribute__((aligned(16))) __bf16 As [2][BM][LDSD];
    __shared__ __attribute__((aligned(16))) __bf16 Bst[2][BN][LDSD];   // [n][k]

    const int tile_m = blockIdx.y * BM;
    const int tile_n = blockIdx.x * BN;
    const int tid  = threadIdx.x;
    const int wid  = tid >> 5;
    const int lane = tid & 31;
    const int wm   = (wid & 3) * 32;    // wave row offset within tile (0..96)
    const int wn   = (wid >> 2) * 32;   // wave col offset within tile (0,32)
    const int r16  = lane & 15;
    const int hf   = lane >> 4;

    // ---- per-thread load coordinates (branchless, clamped) ----
    const int ar   = tid >> 1;             // 0..127  A row within tile
    const int aseg = (tid & 1) * 16;       // 0 / 16  A k-segment (16 floats)
    int agr = tile_m + ar; if (agr >= M) agr = M - 1;

    const int bc = tid & 63;               // 0..63  B column within tile
    const int bk = (tid >> 6) * 8;         // 0,8,16,24  B k-segment
    int bgc = tile_n + bc; if (bgc >= N) bgc = N - 1;

    // running pointers: one 64-bit add per K-step
    const float* aPtr = A + (size_t)agr * lda + aseg;
    const float* wPtr = W + (size_t)bk * ldw + bgc;
    const size_t wStep = (size_t)BK * ldw;

    floatx8 c00 = {}, c01 = {}, c10 = {}, c11 = {};

    float4 a_reg0, a_reg1, a_reg2, a_reg3;
    float  b_reg[8];

    auto load_regs = [&]() {
        const float4* ap = (const float4*)aPtr;
        a_reg0 = ap[0]; a_reg1 = ap[1]; a_reg2 = ap[2]; a_reg3 = ap[3];
#pragma unroll
        for (int j = 0; j < 8; ++j) b_reg[j] = wPtr[(size_t)j * ldw];
        aPtr += BK;
        wPtr += wStep;
    };
    auto store_tile = [&](int buf) {
        *(bf16x8*)&As[buf][ar][aseg]     = pack_bf16x8(a_reg0, a_reg1);
        *(bf16x8*)&As[buf][ar][aseg + 8] = pack_bf16x8(a_reg2, a_reg3);
        bf16x8 bvv;
#pragma unroll
        for (int j = 0; j < 8; ++j) bvv[j] = (__bf16)b_reg[j];
        *(bf16x8*)&Bst[buf][bc][bk] = bvv;
    };

    const int nIter = K / BK;      // K is always a multiple of 32 here
    load_regs();
    store_tile(0);
    __syncthreads();

    int buf = 0;
    for (int it = 0; it < nIter; ++it) {
        const bool more = (it + 1 < nIter);
        if (more) load_regs();     // HBM latency hides under the WMMAs below

        // ---- fragments: pure 16B LDS vector loads ----
        const int arow0 = wm + r16;
        const int arow1 = arow0 + 16;
        bf16x16 a0 = cat16(*(const bf16x8*)&As[buf][arow0][hf * 8],
                           *(const bf16x8*)&As[buf][arow0][16 + hf * 8]);
        bf16x16 a1 = cat16(*(const bf16x8*)&As[buf][arow1][hf * 8],
                           *(const bf16x8*)&As[buf][arow1][16 + hf * 8]);
        const int bcol0 = wn + r16;
        bf16x16 b0 = cat16(*(const bf16x8*)&Bst[buf][bcol0][hf * 8],
                           *(const bf16x8*)&Bst[buf][bcol0][16 + hf * 8]);
        bf16x16 b1 = cat16(*(const bf16x8*)&Bst[buf][bcol0 + 16][hf * 8],
                           *(const bf16x8*)&Bst[buf][bcol0 + 16][16 + hf * 8]);

        c00 = __builtin_amdgcn_wmma_f32_16x16x32_bf16(false, a0, false, b0, (short)0, c00, false, false);
        c01 = __builtin_amdgcn_wmma_f32_16x16x32_bf16(false, a0, false, b1, (short)0, c01, false, false);
        c10 = __builtin_amdgcn_wmma_f32_16x16x32_bf16(false, a1, false, b0, (short)0, c10, false, false);
        c11 = __builtin_amdgcn_wmma_f32_16x16x32_bf16(false, a1, false, b1, (short)0, c11, false, false);

        if (more) store_tile(buf ^ 1);   // other buffer: no conflict with reads above
        __syncthreads();                 // single barrier per K-step
        buf ^= 1;
    }

    // ---- epilogue: bias (+ReLU), store ----
    const int col0 = tile_n + wn + r16;
    const int col1 = col0 + 16;
    const int rb0  = tile_m + wm + hf * 8;
    const int rb1  = rb0 + 16;
    const float bias0 = bias[col0 < N ? col0 : N - 1];
    const float bias1 = bias[col1 < N ? col1 : N - 1];

    auto store_frag = [&](const floatx8& c, int rowbase, int col, float bs) {
        if (col >= N) return;
#pragma unroll
        for (int i = 0; i < 8; ++i) {
            int row = rowbase + i;
            if (row < M) {
                float v = c[i] + bs;
                if (relu && v < 0.f) v = 0.f;
                C[(size_t)row * ldc + col] = v;
            }
        }
    };
    store_frag(c00, rb0, col0, bias0);
    store_frag(c01, rb0, col1, bias1);
    store_frag(c10, rb1, col0, bias0);
    store_frag(c11, rb1, col1, bias1);
}

// =====================================================================
// Self-attention per (b,h): 32 query rows / block, 2 waves, WMMA for
// QK^T and P*V; softmax stats via LDS logits buffer (bf16).
//   Kr : [key][dim]  (row-major)  -> QK^T B-frag = 2x 16B loads
//   Vt : [dim][key]  (transposed) -> P*V  B-frag = 2x 16B loads
// =====================================================================
#define ATT_ROWS 32
#define ATT_TL   928      // keys padded to multiple of 32

__global__ __launch_bounds__(64) void attn_kernel(
    const float* __restrict__ q, const float* __restrict__ k,
    const float* __restrict__ v, float* __restrict__ sa)
{
    __shared__ __attribute__((aligned(16))) __bf16 logits[ATT_ROWS][ATT_TL];  // 59392 B
    __shared__ __attribute__((aligned(16))) __bf16 Kr[32][LDSD];
    __shared__ __attribute__((aligned(16))) __bf16 Vt[HDn][LDSD];
    __shared__ float  rowmax[ATT_ROWS];
    __shared__ float  rowsum[ATT_ROWS];

    const int bh = blockIdx.y;
    const int b  = bh >> 3;
    const int h  = bh & 7;
    const int q0 = blockIdx.x * ATT_ROWS;
    const int tid  = threadIdx.x;
    const int wid  = tid >> 5;
    const int lane = tid & 31;
    const int r16  = lane & 15;
    const int hf   = lane >> 4;
    const float scale = 0.17677669529663687f;   // 1/sqrt(32)

    // ---- per-wave Q fragment (rows q0 + wid*16 .. +15), pre-scaled ----
    int qrow = q0 + wid * 16 + r16; if (qrow >= NQn) qrow = NQn - 1;
    const float4* qbase = (const float4*)(q + ((size_t)b * NQn + qrow) * Dd + h * HDn);
    bf16x16 qa;
    {
        float4 q0v = qbase[hf * 2], q1v = qbase[hf * 2 + 1];         // K hf*8..hf*8+7
        float4 q2v = qbase[4 + hf * 2], q3v = qbase[4 + hf * 2 + 1]; // K 16+hf*8..
        bf16x8 lo, hi;
#pragma unroll
        for (int i = 0; i < 4; ++i) {
            lo[i]     = (__bf16)(((const float*)&q0v)[i] * scale);
            lo[4 + i] = (__bf16)(((const float*)&q1v)[i] * scale);
            hi[i]     = (__bf16)(((const float*)&q2v)[i] * scale);
            hi[4 + i] = (__bf16)(((const float*)&q3v)[i] * scale);
        }
        qa = cat16(lo, hi);
    }

    // per-thread chunk-load coordinates
    const int lkey = tid >> 1;            // 0..31 key within chunk
    const int lseg = (tid & 1) * 16;      // 0 / 16 dim segment

    // ================= pass 1: logits = (Q/sqrt(d)) K^T =================
    for (int kc = 0; kc < ATT_TL; kc += 32) {
        {
            int gk = kc + lkey; if (gk >= NQn) gk = NQn - 1;
            const float4* kp = (const float4*)(k + ((size_t)b * NQn + gk) * Dd + h * HDn + lseg);
            float4 k0v = kp[0], k1v = kp[1], k2v = kp[2], k3v = kp[3];
            *(bf16x8*)&Kr[lkey][lseg]     = pack_bf16x8(k0v, k1v);
            *(bf16x8*)&Kr[lkey][lseg + 8] = pack_bf16x8(k2v, k3v);
        }
        __syncthreads();
#pragma unroll
        for (int nc = 0; nc < 32; nc += 16) {
            const int bcol = nc + r16;
            bf16x16 bk = cat16(*(const bf16x8*)&Kr[bcol][hf * 8],
                               *(const bf16x8*)&Kr[bcol][16 + hf * 8]);
            floatx8 s = {};
            s = __builtin_amdgcn_wmma_f32_16x16x32_bf16(false, qa, false, bk, (short)0, s, false, false);
            int key = kc + nc + r16;
#pragma unroll
            for (int i = 0; i < 8; ++i) {
                int lrow = wid * 16 + hf * 8 + i;
                logits[lrow][key] = (__bf16)((key < NQn) ? s[i] : -1.0e30f);
            }
        }
        __syncthreads();
    }

    // ================= pass 2: row max / sum-exp (8-wide scans) =========
    if (tid < ATT_ROWS) {
        float m = -1.0e30f;
        for (int j = 0; j < ATT_TL; j += 8) {
            bf16x8 l8 = *(const bf16x8*)&logits[tid][j];
#pragma unroll
            for (int i = 0; i < 8; ++i) {
                float x = (float)l8[i];
                m = (x > m) ? x : m;
            }
        }
        float ssum = 0.f;
        for (int j = 0; j < ATT_TL; j += 8) {
            bf16x8 l8 = *(const bf16x8*)&logits[tid][j];
#pragma unroll
            for (int i = 0; i < 8; ++i) ssum += __expf((float)l8[i] - m);
        }
        rowmax[tid] = m;
        rowsum[tid] = ssum + 1e-20f;
    }
    __syncthreads();

    // ================= pass 3: O = softmax(logits) * V ==================
    floatx8 o0 = {}, o1 = {};
    const int prow   = wid * 16 + r16;       // A-frag row for P
    const float pm   = rowmax[prow];
    const float pinv = 1.f / rowsum[prow];
    for (int kc = 0; kc < ATT_TL; kc += 32) {
        {
            int gk = kc + lkey; if (gk >= NQn) gk = NQn - 1;
            const float4* vp = (const float4*)(v + ((size_t)b * NQn + gk) * Dd + h * HDn + lseg);
            float4 v0v = vp[0], v1v = vp[1], v2v = vp[2], v3v = vp[3];
            const float* vf = (const float*)&v0v;
#pragma unroll
            for (int j = 0; j < 4; ++j) Vt[lseg + j][lkey]      = (__bf16)vf[j];
            vf = (const float*)&v1v;
#pragma unroll
            for (int j = 0; j < 4; ++j) Vt[lseg + 4 + j][lkey]  = (__bf16)vf[j];
            vf = (const float*)&v2v;
#pragma unroll
            for (int j = 0; j < 4; ++j) Vt[lseg + 8 + j][lkey]  = (__bf16)vf[j];
            vf = (const float*)&v3v;
#pragma unroll
            for (int j = 0; j < 4; ++j) Vt[lseg + 12 + j][lkey] = (__bf16)vf[j];
        }
        __syncthreads();

        bf16x8 l0 = *(const bf16x8*)&logits[prow][kc + hf * 8];
        bf16x8 l1 = *(const bf16x8*)&logits[prow][kc + 16 + hf * 8];
        bf16x16 pf;
#pragma unroll
        for (int i = 0; i < 8; ++i) {
            pf[i]     = (__bf16)(__expf((float)l0[i] - pm) * pinv);
            pf[8 + i] = (__bf16)(__expf((float)l1[i] - pm) * pinv);
        }
        bf16x16 bv0 = cat16(*(const bf16x8*)&Vt[r16][hf * 8],
                            *(const bf16x8*)&Vt[r16][16 + hf * 8]);
        bf16x16 bv1 = cat16(*(const bf16x8*)&Vt[16 + r16][hf * 8],
                            *(const bf16x8*)&Vt[16 + r16][16 + hf * 8]);
        o0 = __builtin_amdgcn_wmma_f32_16x16x32_bf16(false, pf, false, bv0, (short)0, o0, false, false);
        o1 = __builtin_amdgcn_wmma_f32_16x16x32_bf16(false, pf, false, bv1, (short)0, o1, false, false);
        __syncthreads();
    }

    // ---- store O ----
#pragma unroll
    for (int i = 0; i < 8; ++i) {
        int row = q0 + wid * 16 + hf * 8 + i;
        if (row < NQn) {
            size_t base = ((size_t)b * NQn + row) * Dd + h * HDn;
            sa[base + r16]      = o0[i];
            sa[base + 16 + r16] = o1[i];
        }
    }
}

// =====================================================================
// Elementwise add
// =====================================================================
__global__ __launch_bounds__(256) void elemadd_kernel(
    const float* __restrict__ a, const float* __restrict__ b,
    float* __restrict__ out, int n)
{
    int i = blockIdx.x * 256 + threadIdx.x;
    if (i < n) out[i] = a[i] + b[i];
}

// =====================================================================
// out[row] = LayerNorm(x[row] + res[row]) ; one 256-thread block per row
// =====================================================================
__global__ __launch_bounds__(256) void add_ln_kernel(
    const float* __restrict__ x, const float* __restrict__ res,
    const float* __restrict__ g, const float* __restrict__ bta,
    float* __restrict__ out)
{
    __shared__ float red[8];
    __shared__ float stats[2];
    const int row  = blockIdx.x;
    const int tid  = threadIdx.x;
    const int wid  = tid >> 5;
    const int lane = tid & 31;

    float v = x[(size_t)row * Dd + tid] + res[(size_t)row * Dd + tid];

    float s = v;
#pragma unroll
    for (int o2 = 16; o2 > 0; o2 >>= 1) s += __shfl_xor(s, o2);
    if (lane == 0) red[wid] = s;
    __syncthreads();
    if (tid == 0) {
        float t = 0.f;
        for (int i = 0; i < 8; ++i) t += red[i];
        stats[0] = t * (1.f / Dd);
    }
    __syncthreads();
    float mean = stats[0];
    float d  = v - mean;
    float sq = d * d;
#pragma unroll
    for (int o2 = 16; o2 > 0; o2 >>= 1) sq += __shfl_xor(sq, o2);
    if (lane == 0) red[wid] = sq;
    __syncthreads();
    if (tid == 0) {
        float t = 0.f;
        for (int i = 0; i < 8; ++i) t += red[i];
        stats[1] = t * (1.f / Dd);
    }
    __syncthreads();
    float var = stats[1];
    out[(size_t)row * Dd + tid] = d * rsqrtf(var + 1e-5f) * g[tid] + bta[tid];
}

// =====================================================================
// softmax over last 16 (L*P) of aw ; one thread per (b,q,h)
// =====================================================================
__global__ __launch_bounds__(256) void aw_softmax_kernel(float* __restrict__ aw)
{
    int i = blockIdx.x * 256 + threadIdx.x;
    if (i >= Bb * NQn * Hh) return;
    float* p = aw + (size_t)i * (Ll * Pp);
    float m = -1.0e30f;
#pragma unroll
    for (int j = 0; j < Ll * Pp; ++j) m = p[j] > m ? p[j] : m;
    float s = 0.f;
    float e[Ll * Pp];
#pragma unroll
    for (int j = 0; j < Ll * Pp; ++j) { e[j] = __expf(p[j] - m); s += e[j]; }
    float inv = 1.f / s;
#pragma unroll
    for (int j = 0; j < Ll * Pp; ++j) p[j] = e[j] * inv;
}

// =====================================================================
// Deformable bilinear sampling: one wave per (b,q,h), lane = channel d
// =====================================================================
__global__ __launch_bounds__(256) void deform_kernel(
    const float* __restrict__ value, const float* __restrict__ offv,
    const float* __restrict__ aw,   const float* __restrict__ refp,
    const int* __restrict__ shapes, float* __restrict__ acc)
{
    int gw   = (blockIdx.x * 256 + threadIdx.x) >> 5;
    int lane = threadIdx.x & 31;
    if (gw >= Bb * NQn * Hh) return;
    int h = gw & 7;
    int qq = (gw >> 3) % NQn;
    int b = gw / (NQn * Hh);

    float a = 0.f;
    int start = 0;
    for (int l = 0; l < Ll; ++l) {
        int Hl = shapes[l * 2 + 0];
        int Wl = shapes[l * 2 + 1];
        float rx = refp[(((size_t)b * NQn + qq) * Ll + l) * 2 + 0];
        float ry = refp[(((size_t)b * NQn + qq) * Ll + l) * 2 + 1];
        for (int p = 0; p < Pp; ++p) {
            size_t oi = ((((size_t)b * NQn + qq) * Hh + h) * Ll + l) * Pp + p;
            float ox = offv[oi * 2 + 0];
            float oy = offv[oi * 2 + 1];
            float x = (rx + ox / (float)Wl) * (float)Wl - 0.5f;
            float y = (ry + oy / (float)Hl) * (float)Hl - 0.5f;
            float x0 = floorf(x), y0 = floorf(y);
            float wx = x - x0, wy = y - y0;
            int x0i = (int)x0, y0i = (int)y0;

            auto tap = [&](int xi, int yi) -> float {
                if (xi < 0 || xi >= Wl || yi < 0 || yi >= Hl) return 0.f;
                return value[((size_t)b * Ss + start + (size_t)yi * Wl + xi) * Dd + h * HDn + lane];
            };
            float samp = tap(x0i,     y0i)     * (1.f - wx) * (1.f - wy)
                       + tap(x0i + 1, y0i)     * wx         * (1.f - wy)
                       + tap(x0i,     y0i + 1) * (1.f - wx) * wy
                       + tap(x0i + 1, y0i + 1) * wx         * wy;
            a += samp * aw[oi];
        }
        start += Hl * Wl;
    }
    acc[((size_t)b * NQn + qq) * Dd + h * HDn + lane] = a;
}

// =====================================================================
// Host launcher
// =====================================================================
extern "C" void kernel_launch(void* const* d_in, const int* in_sizes, int n_in,
                              void* d_out, int out_size, void* d_ws, size_t ws_size,
                              hipStream_t stream)
{
    (void)in_sizes; (void)n_in; (void)out_size; (void)ws_size;

    const float* tgt        = (const float*)d_in[0];
    const float* query_pos  = (const float*)d_in[1];
    const float* refp       = (const float*)d_in[2];
    const float* memory     = (const float*)d_in[3];
    const float* in_proj_w  = (const float*)d_in[4];
    const float* in_proj_b  = (const float*)d_in[5];
    const float* out_proj_w = (const float*)d_in[6];
    const float* out_proj_b = (const float*)d_in[7];
    const float* ln1_g      = (const float*)d_in[8];
    const float* ln1_b      = (const float*)d_in[9];
    const float* ln2_g      = (const float*)d_in[10];
    const float* ln2_b      = (const float*)d_in[11];
    const float* ln3_g      = (const float*)d_in[12];
    const float* ln3_b      = (const float*)d_in[13];
    const float* vproj_w    = (const float*)d_in[14];
    const float* vproj_b    = (const float*)d_in[15];
    const float* off_w      = (const float*)d_in[16];
    const float* off_b      = (const float*)d_in[17];
    const float* aw_w       = (const float*)d_in[18];
    const float* aw_b       = (const float*)d_in[19];
    const float* oproj_w    = (const float*)d_in[20];
    const float* oproj_b    = (const float*)d_in[21];
    const float* lin1_w     = (const float*)d_in[22];
    const float* lin1_b     = (const float*)d_in[23];
    const float* lin2_w     = (const float*)d_in[24];
    const float* lin2_b     = (const float*)d_in[25];
    const int*   shapes     = (const int*)d_in[26];

    float* wsf = (float*)d_ws;
    size_t woff = 0;
    auto take = [&](size_t n) { float* p = wsf + woff; woff += n; return p; };

    const size_t MQ = (size_t)Bb * NQn;    // 3600
    const size_t MS = (size_t)Bb * Ss;     // 87040

    float* qp   = take(MQ * Dd);
    float* qb   = take(MQ * Dd);
    float* kb   = take(MQ * Dd);
    float* vb   = take(MQ * Dd);
    float* sa   = take(MQ * Dd);
    float* tmp1 = take(MQ * Dd);
    float* tgt2 = take(MQ * Dd);
    float* qry  = take(MQ * Dd);
    float* offb = take(MQ * Dd);
    float* awb  = take(MQ * (Hh * Ll * Pp));
    float* accb = take(MQ * Dd);
    float* tmp2 = take(MQ * Dd);
    float* tgt3 = take(MQ * Dd);
    float* ff1  = take(MQ * DFFn);
    float* ff2  = take(MQ * Dd);
    float* valb = take(MS * Dd);

    const int nAdd = (int)(MQ * Dd);
    const dim3 gAdd((nAdd + 255) / 256);
    const dim3 gN256(Dd / BN, (unsigned)((MQ + BM - 1) / BM));

    // qp = tgt + query_pos
    elemadd_kernel<<<gAdd, 256, 0, stream>>>(tgt, query_pos, qp, nAdd);

    // q / k / v projections (column slices of in_proj_w, ld = 3D)
    gemm_bf16_wmma<<<gN256, 256, 0, stream>>>(qp,  (int)MQ, Dd, Dd, in_proj_w,          3 * Dd, in_proj_b,          qb, Dd, Dd, 0);
    gemm_bf16_wmma<<<gN256, 256, 0, stream>>>(qp,  (int)MQ, Dd, Dd, in_proj_w + Dd,     3 * Dd, in_proj_b + Dd,     kb, Dd, Dd, 0);
    gemm_bf16_wmma<<<gN256, 256, 0, stream>>>(tgt, (int)MQ, Dd, Dd, in_proj_w + 2 * Dd, 3 * Dd, in_proj_b + 2 * Dd, vb, Dd, Dd, 0);

    // self-attention (WMMA QK^T + PV)
    attn_kernel<<<dim3((NQn + ATT_ROWS - 1) / ATT_ROWS, Bb * Hh), 64, 0, stream>>>(qb, kb, vb, sa);

    // out-proj + residual LN (ln2)
    gemm_bf16_wmma<<<gN256, 256, 0, stream>>>(sa, (int)MQ, Dd, Dd, out_proj_w, Dd, out_proj_b, tmp1, Dd, Dd, 0);
    add_ln_kernel<<<(unsigned)MQ, 256, 0, stream>>>(tmp1, tgt, ln2_g, ln2_b, tgt2);

    // query = tgt2 + query_pos
    elemadd_kernel<<<gAdd, 256, 0, stream>>>(tgt2, query_pos, qry, nAdd);

    // sampling offsets (N = H*L*P*2 = 256) and attention weights (N = 128)
    gemm_bf16_wmma<<<gN256, 256, 0, stream>>>(qry, (int)MQ, Dd, Dd, off_w, Hh * Ll * Pp * 2, off_b, offb, Hh * Ll * Pp * 2, Hh * Ll * Pp * 2, 0);
    gemm_bf16_wmma<<<dim3((Hh * Ll * Pp) / BN, (unsigned)((MQ + BM - 1) / BM)), 256, 0, stream>>>(
        qry, (int)MQ, Dd, Dd, aw_w, Hh * Ll * Pp, aw_b, awb, Hh * Ll * Pp, Hh * Ll * Pp, 0);
    aw_softmax_kernel<<<(Bb * NQn * Hh + 255) / 256, 256, 0, stream>>>(awb);

    // value projection (the big GEMM: 87040 x 256 x 256)
    gemm_bf16_wmma<<<dim3(Dd / BN, (unsigned)((MS + BM - 1) / BM)), 256, 0, stream>>>(
        memory, (int)MS, Dd, Dd, vproj_w, Dd, vproj_b, valb, Dd, Dd, 0);

    // deformable bilinear sampling
    deform_kernel<<<(Bb * NQn * Hh * 32 + 255) / 256, 256, 0, stream>>>(valb, offb, awb, refp, shapes, accb);

    // output projection + residual LN (ln1)
    gemm_bf16_wmma<<<gN256, 256, 0, stream>>>(accb, (int)MQ, Dd, Dd, oproj_w, Dd, oproj_b, tmp2, Dd, Dd, 0);
    add_ln_kernel<<<(unsigned)MQ, 256, 0, stream>>>(tmp2, tgt2, ln1_g, ln1_b, tgt3);

    // FFN: 256 -> 1024 (ReLU) -> 256, residual LN (ln3) to output
    gemm_bf16_wmma<<<dim3(DFFn / BN, (unsigned)((MQ + BM - 1) / BM)), 256, 0, stream>>>(
        tgt3, (int)MQ, Dd, Dd, lin1_w, DFFn, lin1_b, ff1, DFFn, DFFn, 1);
    gemm_bf16_wmma<<<gN256, 256, 0, stream>>>(ff1, (int)MQ, DFFn, DFFn, lin2_w, Dd, lin2_b, ff2, Dd, Dd, 0);
    add_ln_kernel<<<(unsigned)MQ, 256, 0, stream>>>(ff2, tgt3, ln3_g, ln3_b, (float*)d_out);
}
